// ThickConv2d_18416819765627
// MI455X (gfx1250) — compile-verified, matrix operand order checked by reading, not run on hardware
//
#include <hip/hip_runtime.h>
#include <hip/hip_bf16.h>

// ---------------------------------------------------------------------------
// ThickConv2d fused as a bf16 WMMA implicit GEMM (MI455X / gfx1250, wave32).
//   Stage-1 conv == GEMM: A[M=8192, K=1152] = w1 reshaped, B[K=1152, N=8192]
//   = im2col(x) (N = b*1024 + y*32 + x).  Epilogue fuses bias + LeakyReLU +
//   per-o contraction over 64 mid-channels (w2, b2) so h never hits HBM.
//
// Wave tile 64(M) x 64(N): 16 accumulators; per K-step 8 fragment loads feed
// 16 v_wmma_f32_16x16x32_bf16 -> 0.5 KB operand traffic per WMMA (2x better
// than a 16x128 tile), keeping the tensor pipe fed from the WGP cache.
// ---------------------------------------------------------------------------

typedef __attribute__((ext_vector_type(16))) __bf16 v16bf;
typedef __attribute__((ext_vector_type(8)))  float  v8f;

#define IN_CH  128
#define MID_CH 64
#define OUT_CH 128
#define KDIM   1152            // IN_CH * 3 * 3
#define KT     36              // KDIM / 32
#define MDIM   8192            // OUT_CH * MID_CH
#define NDIM   8192            // 8 * 32 * 32
#define MT     (MDIM / 16)     // 512
#define NT     (NDIM / 16)     // 512
#define PACK_THREADS (MT * KT * 32)   // one thread per (tile, lane) -> 16 bf16

// fragment element mapping (ISA 7.12.2, 16-bit A 16x32 / B 32x16):
//   k = (e>>3)*16 + (lane>>4)*8 + (e&7)
__device__ __forceinline__ int frag_k(int lane, int e) {
    return ((e >> 3) << 4) + ((lane >> 4) << 3) + (e & 7);
}

// Pack w1 [8192][1152] fp32 -> bf16 fragments: Apk[((mt*KT+kt)*32+lane)*16+e]
__global__ __launch_bounds__(256) void pack_a_kernel(const float* __restrict__ w1,
                                                     __bf16* __restrict__ Apk) {
    int t = blockIdx.x * 256 + threadIdx.x;
    if (t >= PACK_THREADS) return;
    int lane = t & 31;
    int kt   = (t >> 5) % KT;
    int mt   = (t >> 5) / KT;
    int m    = mt * 16 + (lane & 15);
    const float* src = w1 + (size_t)m * KDIM + kt * 32;
    v16bf v;
#pragma unroll
    for (int e = 0; e < 16; ++e)
        v[e] = (__bf16)src[frag_k(lane, e)];
    *(v16bf*)(Apk + (size_t)t * 16) = v;
}

// Pack im2col(x) -> bf16 fragments: Bpk[((nt*KT+kt)*32+lane)*16+e]
__global__ __launch_bounds__(256) void pack_b_kernel(const float* __restrict__ x,
                                                     __bf16* __restrict__ Bpk) {
    int t = blockIdx.x * 256 + threadIdx.x;
    if (t >= PACK_THREADS) return;
    int lane = t & 31;
    int kt   = (t >> 5) % KT;
    int nt   = (t >> 5) / KT;
    int n    = nt * 16 + (lane & 15);
    int b    = n >> 10;
    int y    = (n >> 5) & 31;
    int xx   = n & 31;
    v16bf v;
#pragma unroll
    for (int e = 0; e < 16; ++e) {
        int k  = kt * 32 + frag_k(lane, e);      // k = c*9 + ky*3 + kx
        int c  = k / 9;
        int r  = k - c * 9;
        int ky = r / 3;
        int kx = r - ky * 3;
        int iy = y + ky - 1;
        int ix = xx + kx - 1;
        float val = 0.0f;
        if ((unsigned)iy < 32u && (unsigned)ix < 32u)
            val = x[(((size_t)b * IN_CH + c) * 32 + iy) * 32 + ix];
        v[e] = (__bf16)val;
    }
    *(v16bf*)(Bpk + (size_t)t * 16) = v;
}

// One block = one output channel o (64 M rows) x 512 N columns.
// 8 waves, each computing the FULL 64 M rows x 64 N columns:
//   acc[wm][t] : wm = M-subtile 0..3, t = N-subtile 0..3  (16 x v8f).
// Epilogue reduces all 64 mid-channels with only 2 LDS adds per column.
__global__ __launch_bounds__(256) void thickconv_wmma_kernel(
        const __bf16* __restrict__ Apk, const __bf16* __restrict__ Bpk,
        const float* __restrict__ b1, const float* __restrict__ w2,
        const float* __restrict__ b2, float* __restrict__ out) {
    __shared__ float red[512];
    const int tid  = threadIdx.x;
    const int o    = blockIdx.y;          // 0..127
    const int ct   = blockIdx.x;          // 0..15 (column tile of 512)
    red[tid]       = 0.0f;
    red[tid + 256] = 0.0f;
    __syncthreads();

    const int lane = tid & 31;
    const int wave = tid >> 5;            // 0..7 -> columns wave*64 .. +63
    const int nt0  = ct * 32 + wave * 4;  // first of 4 N-subtiles

    const v16bf* __restrict__ Abase = (const v16bf*)Apk;
    const v16bf* __restrict__ Bbase = (const v16bf*)Bpk;

    v8f acc[4][4];
#pragma unroll
    for (int wm = 0; wm < 4; ++wm)
#pragma unroll
        for (int t = 0; t < 4; ++t) acc[wm][t] = (v8f)(0.0f);

    for (int kt = 0; kt < KT; ++kt) {
        // 4 A fragments (one per M-subtile of channel o), shared by all waves
        v16bf a[4];
#pragma unroll
        for (int wm = 0; wm < 4; ++wm)
            a[wm] = Abase[((size_t)(o * 4 + wm) * KT + kt) * 32 + lane];
        if (kt + 1 < KT) {
            __builtin_prefetch((const void*)(Abase +
                ((size_t)(o * 4) * KT + (kt + 1)) * 32 + lane), 0, 0);
            __builtin_prefetch((const void*)(Bbase +
                ((size_t)nt0 * KT + (kt + 1)) * 32 + lane), 0, 0);
        }
#pragma unroll
        for (int t = 0; t < 4; ++t) {
            v16bf bfr = Bbase[((size_t)(nt0 + t) * KT + kt) * 32 + lane];
#pragma unroll
            for (int wm = 0; wm < 4; ++wm) {
                acc[wm][t] = __builtin_amdgcn_wmma_f32_16x16x32_bf16(
                    /*neg_a=*/false, a[wm], /*neg_b=*/false, bfr,
                    /*c_mod=*/(short)0, acc[wm][t],
                    /*reuse_a=*/false, /*reuse_b=*/false);
            }
        }
    }

    // Fused epilogue: +b1, LeakyReLU(0.1), * w2[o,m], reduce over 64 mids.
    // D layout: vgpr v -> row (wm*16 + 8*(lane>>4) + v), lane&15 -> column.
    // Each lane covers 32 of the 64 mids; lane^16 covers the other 32.
    const int g = (lane >> 4) << 3;
    float psum[4] = {0.0f, 0.0f, 0.0f, 0.0f};
#pragma unroll
    for (int wm = 0; wm < 4; ++wm) {
        const int mbase = o * MID_CH + wm * 16 + g;
        float bv[8], cv[8];
#pragma unroll
        for (int v = 0; v < 8; ++v) {
            bv[v] = b1[mbase + v];
            cv[v] = w2[mbase + v];
        }
#pragma unroll
        for (int t = 0; t < 4; ++t) {
            float sum = psum[t];
#pragma unroll
            for (int v = 0; v < 8; ++v) {
                float hv = acc[wm][t][v] + bv[v];
                hv = (hv >= 0.0f) ? hv : 0.1f * hv;   // LeakyReLU(0.1)
                sum += cv[v] * hv;
            }
            psum[t] = sum;
        }
    }
#pragma unroll
    for (int t = 0; t < 4; ++t) {
        const int lcol = (wave * 4 + t) * 16 + (lane & 15);   // 0..511
        atomicAdd(&red[lcol], psum[t]);                        // ds_add_f32 x2/col
    }
    __syncthreads();

    // Write out[b, o, y, x]; column n = ct*512 + i = b*1024 + y*32 + x.
#pragma unroll
    for (int i = tid; i < 512; i += 256) {
        const int col = ct * 512 + i;
        const float val = red[i] + b2[o];
        out[(size_t)(col >> 10) * (OUT_CH * 1024) + (size_t)o * 1024 +
            (col & 1023)] = val;
    }
}

extern "C" void kernel_launch(void* const* d_in, const int* in_sizes, int n_in,
                              void* d_out, int out_size, void* d_ws, size_t ws_size,
                              hipStream_t stream) {
    const float* x  = (const float*)d_in[0];
    const float* w1 = (const float*)d_in[1];
    const float* b1 = (const float*)d_in[2];
    const float* w2 = (const float*)d_in[3];
    const float* b2 = (const float*)d_in[4];
    float* out = (float*)d_out;

    __bf16* Apk = (__bf16*)d_ws;                       // 8192*1152*2 = 18.9 MB
    __bf16* Bpk = Apk + (size_t)MDIM * KDIM;           // +18.9 MB

    const int packBlocks = (PACK_THREADS + 255) / 256; // 2304
    pack_a_kernel<<<packBlocks, 256, 0, stream>>>(w1, Apk);
    pack_b_kernel<<<packBlocks, 256, 0, stream>>>(x, Bpk);

    dim3 grid(NDIM / 512, OUT_CH);                     // (16, 128)
    thickconv_wmma_kernel<<<grid, 256, 0, stream>>>(Apk, Bpk, b1, w2, b2, out);
}
